// NonLocal2D_14482629722201
// MI455X (gfx1250) — matrix operand order repaired and checked by reading, not compiled, not run on hardware
//
#include <hip/hip_runtime.h>
#include <hip/hip_bf16.h>
#include <stdint.h>

// NonLocal2D attention block, CDNA5 (gfx1250), wave32 + WMMA f16->f32.
// Pipeline:
//   k_wcvt      : Wg/Wt/Wp/Wo f32 -> f16 once (kills redundant per-wave cvts)
//   k_transpose : x (n,c,p) f32 -> xf (n,p,c) f16              (LDS tiled)
//   k_proj      : xf @ {Wg,Wt,Wp}^T -> gT (n,c,p), theta/phi (n,p,c)  [WMMA]
//   k_attn      : flash attention, 4-wave WG, async double-buffered
//                 phi/g key blocks in LDS (ASYNCcnt pipeline)         [WMMA]
//   k_out       : y @ Wo^T + bo + x -> out (n,c,h,w) f32              [WMMA]
// Workspace: 5 * (4*4096*256) f16 tensors + 4*256*256 f16 weights = 40.5 MB
// (all attention operands L2-resident on the 192 MB L2).

constexpr int NB = 4;
constexpr int C  = 256;
constexpr int NP = 64 * 64;   // 4096 spatial positions

typedef __attribute__((ext_vector_type(16))) _Float16 v16h;
typedef __attribute__((ext_vector_type(8)))  float    v8f;

__device__ __forceinline__ v8f wmma16(v16h a, v16h b, v8f c) {
  return __builtin_amdgcn_wmma_f32_16x16x32_f16(false, a, false, b, (short)0, c,
                                                false, false);
}

// A fragment, 16-bit 16x32 (MxK): lane holds row M=lane&15; halves[0..15]
// = K = lhi*16 + {0..15}, contiguous.
// `base` already includes the lhi*16 half offset.
__device__ __forceinline__ v16h load_a_frag(const _Float16* base) {
  v16h r;
  __builtin_memcpy(&r, base, 32);
  return r;
}

// B fragment, 16-bit 32x16 (KxN): lane holds column N=lane&15; halves[0..15]
// = K = lhi*16 + {0..15}, contiguous. `base` includes the lhi*16 offset.
__device__ __forceinline__ v16h load_b_frag(const _Float16* base) {
  v16h r;
  __builtin_memcpy(&r, base, 32);
  return r;
}

// gfx1250 async global->LDS copy, 16B per lane (tracked by ASYNCcnt).
// Generic LDS pointers carry the wave-relative LDS byte address in [31:0].
__device__ __forceinline__ void async_copy_b128(const _Float16* g,
                                                _Float16* l) {
  const unsigned lds = (unsigned)(uintptr_t)l;
  asm volatile("global_load_async_to_lds_b128 %0, %1, off"
               :: "v"(lds), "v"(g)
               : "memory");
}

// Intra-wave LDS producer->consumer fence (per-wave LDS regions only):
// LDS pipe is in-order per wave; this orders VALU reads after our DS ops.
__device__ __forceinline__ void wave_lds_fence() {
  asm volatile("s_wait_dscnt 0x0" ::: "memory");
}

// ---------------------------------------------------------------------------
// Kernel W: one-shot f32 -> f16 conversion of the four weight matrices.
// ---------------------------------------------------------------------------
__global__ __launch_bounds__(256) void k_wcvt(const float* __restrict__ Wg,
                                              const float* __restrict__ Wt,
                                              const float* __restrict__ Wp,
                                              const float* __restrict__ Wo,
                                              _Float16* __restrict__ w16) {
  const int i = blockIdx.x * 256 + threadIdx.x;          // 0..65535
  w16[0 * C * C + i] = (_Float16)Wg[i];
  w16[1 * C * C + i] = (_Float16)Wt[i];
  w16[2 * C * C + i] = (_Float16)Wp[i];
  w16[3 * C * C + i] = (_Float16)Wo[i];
}

// ---------------------------------------------------------------------------
// Kernel 0: transpose + f32->f16 convert: xf[n][p][c] = (f16) x[n][c][p]
// ---------------------------------------------------------------------------
__global__ __launch_bounds__(256) void k_transpose(const float* __restrict__ x,
                                                   _Float16* __restrict__ xf) {
  __shared__ float tile[32][33];
  const int n  = blockIdx.z;
  const int p0 = blockIdx.x * 32;
  const int c0 = blockIdx.y * 32;
  const int tx = threadIdx.x;
  const int ty = threadIdx.y;
#pragma unroll
  for (int i = 0; i < 32; i += 8)
    tile[ty + i][tx] = x[((size_t)n * C + (c0 + ty + i)) * NP + p0 + tx];
  __syncthreads();
#pragma unroll
  for (int i = 0; i < 32; i += 8)
    xf[((size_t)n * NP + (p0 + ty + i)) * C + c0 + tx] =
        (_Float16)tile[tx][ty + i];
}

// ---------------------------------------------------------------------------
// Kernel 1: fused triple projection (g/theta/phi share the A fragment).
// ---------------------------------------------------------------------------
__global__ __launch_bounds__(32) void k_proj(
    const _Float16* __restrict__ xf, const _Float16* __restrict__ w16,
    const float* __restrict__ bg, const float* __restrict__ bt,
    const float* __restrict__ bp,
    _Float16* __restrict__ gT, _Float16* __restrict__ th,
    _Float16* __restrict__ ph) {
  const int lane = threadIdx.x;
  const int lrow = lane & 15;
  const int lhi  = lane >> 4;
  const int pb = blockIdx.x * 16;
  const int cb = blockIdx.y * 16;
  const int n  = blockIdx.z;
  const int co = cb + lrow;

  const _Float16* arow = xf + ((size_t)n * NP + pb + lrow) * C + lhi * 16;
  const _Float16* wg = w16 + 0 * C * C + (size_t)co * C + lhi * 16;
  const _Float16* wt = w16 + 1 * C * C + (size_t)co * C + lhi * 16;
  const _Float16* wp = w16 + 2 * C * C + (size_t)co * C + lhi * 16;

  v8f ag = {0, 0, 0, 0, 0, 0, 0, 0};
  v8f at = {0, 0, 0, 0, 0, 0, 0, 0};
  v8f ap = {0, 0, 0, 0, 0, 0, 0, 0};
#pragma unroll
  for (int ch = 0; ch < 8; ++ch) {
    v16h A = load_a_frag(arow + ch * 32);
    ag = wmma16(A, load_b_frag(wg + ch * 32), ag);
    at = wmma16(A, load_b_frag(wt + ch * 32), at);
    ap = wmma16(A, load_b_frag(wp + ch * 32), ap);
  }
  const float bgv = bg[co], btv = bt[co], bpv = bp[co];
#pragma unroll
  for (int j = 0; j < 8; ++j) {
    const int row = pb + j + 8 * lhi;                  // D: M = j + 8*lhi
    const size_t ridx = ((size_t)n * NP + row) * C + co;
    th[ridx] = (_Float16)(at[j] + btv);
    ph[ridx] = (_Float16)(ap[j] + bpv);
    gT[((size_t)n * C + co) * NP + row] = (_Float16)(ag[j] + bgv);
  }
}

// ---------------------------------------------------------------------------
// Kernel 2: flash attention. 4 waves / WG; each wave owns 16 queries x all
// 256 channels. phi/g 32-key blocks are async double-buffered into LDS and
// shared by all 4 waves (4x traffic cut vs per-wave global reads).
// ---------------------------------------------------------------------------

// Stage one 32-key block: this wave's share = 8 phi rows + 64 g rows (16 async
// b128 copies -> the s_wait_asynccnt 0x10 in the main loop matches this).
__device__ __forceinline__ void stage_kblock(const _Float16* __restrict__ phN,
                                             const _Float16* __restrict__ gN,
                                             int kb, _Float16* phDst,
                                             _Float16* gDst, int w, int lane) {
#pragma unroll
  for (int i = 0; i < 8; ++i) {                       // phi rows (keys)
    const int r = w * 8 + i;                          // 0..31
    async_copy_b128(phN + (size_t)(kb + r) * C + lane * 8,
                    phDst + r * 256 + lane * 8);
  }
#pragma unroll
  for (int i = 0; i < 8; ++i) {                       // g rows (channels)
    const int cc = w * 64 + i * 8 + (lane >> 2);      // 0..255
    const int o  = (lane & 3) * 8;                    // half offset in row
    async_copy_b128(gN + (size_t)cc * NP + kb + o, gDst + cc * 32 + o);
  }
}

__global__ __launch_bounds__(128) void k_attn(const _Float16* __restrict__ th,
                                              const _Float16* __restrict__ ph,
                                              const _Float16* __restrict__ gT,
                                              _Float16* __restrict__ yF) {
  __shared__ _Float16 thL[64 * 256];        // 32 KB: 64 query rows
  __shared__ _Float16 phiL[2][32 * 256];    // 32 KB: double-buffered keys
  __shared__ _Float16 gL[2][256 * 32];      // 32 KB: double-buffered values
  __shared__ float    Sl[4][16 * 32];       // per-wave logits
  __shared__ _Float16 Pl[4][16 * 32];       // per-wave probs
  __shared__ float    mS[4][16], lS[4][16], scS[4][16];

  const int tid  = threadIdx.x;
  const int w    = tid >> 5;
  const int lane = tid & 31;
  const int lrow = lane & 15;
  const int lhi  = lane >> 4;
  const int qb   = blockIdx.x * 64;
  const int n    = blockIdx.y;

  const _Float16* thN = th + ((size_t)n * NP + qb) * C;
  const _Float16* phN = ph + (size_t)n * NP * C;
  const _Float16* gN  = gT + (size_t)n * C * NP;

  // Prologue: async-stage this wave's theta tile + first key block.
#pragma unroll
  for (int i = 0; i < 16; ++i) {
    const int r = w * 16 + i;
    async_copy_b128(thN + (size_t)r * C + lane * 8, &thL[r * 256 + lane * 8]);
  }
  stage_kblock(phN, gN, 0, &phiL[0][0], &gL[0][0], w, lane);
  if (lane < 16) { mS[w][lane] = -3.0e38f; lS[w][lane] = 0.f; }
  asm volatile("s_wait_asynccnt 0x0" ::: "memory");
  __syncthreads();

  v8f acc[16];
#pragma unroll
  for (int t = 0; t < 16; ++t) acc[t] = (v8f){0, 0, 0, 0, 0, 0, 0, 0};

  float* SlW = Sl[w];
  _Float16* PlW = Pl[w];

  constexpr int NS = NP / 32;               // 128 key steps
  for (int s = 0; s < NS; ++s) {
    const int cur = s & 1;
    if (s + 1 < NS) {
      // Kick the next block's 16 copies, then wait until only those 16 are
      // outstanding (ASYNCcnt completes in order -> current buffer is done).
      stage_kblock(phN, gN, (s + 1) * 32, &phiL[cur ^ 1][0], &gL[cur ^ 1][0],
                   w, lane);
      asm volatile("s_wait_asynccnt 0x10" ::: "memory");
    } else {
      asm volatile("s_wait_asynccnt 0x0" ::: "memory");
    }
    __syncthreads();                        // publish current buffer to all waves

    const _Float16* phB = &phiL[cur][0];
    const _Float16* gB  = &gL[cur][0];

    // ---- logits S(16q x 32k) = theta . phi^T ----
    v8f s0 = {0, 0, 0, 0, 0, 0, 0, 0};
    v8f s1 = {0, 0, 0, 0, 0, 0, 0, 0};
#pragma unroll
    for (int ch = 0; ch < 8; ++ch) {
      v16h A  = load_a_frag(&thL[(w * 16 + lrow) * 256 + ch * 32 + lhi * 16]);
      v16h B0 = load_b_frag(&phB[lrow * 256 + ch * 32 + lhi * 16]);
      v16h B1 = load_b_frag(&phB[(16 + lrow) * 256 + ch * 32 + lhi * 16]);
      s0 = wmma16(A, B0, s0);
      s1 = wmma16(A, B1, s1);
    }
#pragma unroll
    for (int j = 0; j < 8; ++j) {
      SlW[(j + 8 * lhi) * 32 + lrow]      = s0[j];
      SlW[(j + 8 * lhi) * 32 + 16 + lrow] = s1[j];
    }
    wave_lds_fence();

    // ---- online softmax (one lane per query row; per-wave LDS region) ----
    if (lane < 16) {
      const int r = lane;
      const float mo = mS[w][r], lo = lS[w][r];
      float mx = mo;
#pragma unroll
      for (int c2 = 0; c2 < 32; ++c2) mx = fmaxf(mx, SlW[r * 32 + c2]);
      float sum = 0.f;
#pragma unroll
      for (int c2 = 0; c2 < 32; ++c2) {
        const float e = __expf(SlW[r * 32 + c2] - mx);
        PlW[r * 32 + c2] = (_Float16)e;
        sum += e;
      }
      const float sc = __expf(mo - mx);
      mS[w][r] = mx;
      lS[w][r] = lo * sc + sum;
      scS[w][r] = sc;
    }
    wave_lds_fence();

    // ---- rescale accumulators, acc += P(16x32) . g(32x16) per c-tile ----
    float f0[8];
#pragma unroll
    for (int j = 0; j < 8; ++j) f0[j] = scS[w][j + 8 * lhi];
#pragma unroll
    for (int t = 0; t < 16; ++t)
#pragma unroll
      for (int j = 0; j < 8; ++j) acc[t][j] *= f0[j];

    v16h Pf = load_a_frag(&PlW[lrow * 32 + lhi * 16]);
#pragma unroll
    for (int t = 0; t < 16; ++t) {
      v16h G = load_b_frag(&gB[(t * 16 + lrow) * 32 + lhi * 16]);
      acc[t] = wmma16(Pf, G, acc[t]);
    }
    __syncthreads();   // all waves done reading buf[cur] before it is refilled
  }

  // ---- normalize and emit y (f16, row-major over channels) ----
  if (lane < 16) scS[w][lane] = 1.f / lS[w][lane];
  wave_lds_fence();
  float inv[8];
#pragma unroll
  for (int j = 0; j < 8; ++j) inv[j] = scS[w][j + 8 * lhi];
#pragma unroll
  for (int t = 0; t < 16; ++t)
#pragma unroll
    for (int j = 0; j < 8; ++j) {
      const int row = qb + w * 16 + j + 8 * lhi;
      yF[((size_t)n * NP + row) * C + t * 16 + lrow] =
          (_Float16)(acc[t][j] * inv[j]);
    }
}

// ---------------------------------------------------------------------------
// Kernel 3: out = x + y @ Wo^T + bo, written fp32 in NCHW layout.
// ---------------------------------------------------------------------------
__global__ __launch_bounds__(32) void k_out(const _Float16* __restrict__ yF,
                                            const _Float16* __restrict__ w16,
                                            const float* __restrict__ bo,
                                            const float* __restrict__ x,
                                            float* __restrict__ out) {
  const int lane = threadIdx.x;
  const int lrow = lane & 15;
  const int lhi  = lane >> 4;
  const int pb = blockIdx.x * 16;
  const int cb = blockIdx.y * 16;
  const int n  = blockIdx.z;
  const int co = cb + lrow;

  const _Float16* arow = yF + ((size_t)n * NP + pb + lrow) * C + lhi * 16;
  const _Float16* wo = w16 + 3 * C * C + (size_t)co * C + lhi * 16;

  v8f acc = {0, 0, 0, 0, 0, 0, 0, 0};
#pragma unroll
  for (int ch = 0; ch < 8; ++ch) {
    v16h A = load_a_frag(arow + ch * 32);
    acc = wmma16(A, load_b_frag(wo + ch * 32), acc);
  }
  const float bov = bo[co];
#pragma unroll
  for (int j = 0; j < 8; ++j) {
    const int row = pb + j + 8 * lhi;
    const size_t idx = ((size_t)n * C + co) * NP + row;   // NCHW
    out[idx] = x[idx] + acc[j] + bov;
  }
}

// ---------------------------------------------------------------------------
extern "C" void kernel_launch(void* const* d_in, const int* in_sizes, int n_in,
                              void* d_out, int out_size, void* d_ws,
                              size_t ws_size, hipStream_t stream) {
  (void)in_sizes; (void)n_in; (void)out_size; (void)ws_size;
  const float* x  = (const float*)d_in[0];
  const float* Wg = (const float*)d_in[1];
  const float* bg = (const float*)d_in[2];
  const float* Wt = (const float*)d_in[3];
  const float* bt = (const float*)d_in[4];
  const float* Wp = (const float*)d_in[5];
  const float* bp = (const float*)d_in[6];
  const float* Wo = (const float*)d_in[7];
  const float* bo = (const float*)d_in[8];
  float* out = (float*)d_out;

  const size_t NE = (size_t)NB * NP * C;          // 4,194,304 elements
  _Float16* xf  = (_Float16*)d_ws;                // 5*NE*2 + 512KB = 40.5 MB
  _Float16* th  = xf + NE;
  _Float16* ph  = th + NE;
  _Float16* gT  = ph + NE;
  _Float16* yF  = gT + NE;
  _Float16* w16 = yF + NE;                        // 4 * 256*256 f16

  k_wcvt<<<dim3(C * C / 256, 1, 1), dim3(256, 1, 1), 0, stream>>>(Wg, Wt, Wp,
                                                                  Wo, w16);
  k_transpose<<<dim3(NP / 32, C / 32, NB), dim3(32, 8, 1), 0, stream>>>(x, xf);
  k_proj<<<dim3(NP / 16, C / 16, NB), dim3(32, 1, 1), 0, stream>>>(
      xf, w16, bg, bt, bp, gT, th, ph);
  k_attn<<<dim3(NP / 64, NB, 1), dim3(128, 1, 1), 0, stream>>>(th, ph, gT, yF);
  k_out<<<dim3(NP / 16, C / 16, NB), dim3(32, 1, 1), 0, stream>>>(yF, w16, bo,
                                                                  x, out);
}
